// PrototypeEmotionModel_12000138625284
// MI455X (gfx1250) — matrix operand order repaired and checked by reading, not compiled, not run on hardware
//
#include <hip/hip_runtime.h>
#include <math.h>
#include <stdint.h>
#include <stddef.h>

// ============================================================================
// MI455X (gfx1250) implementation.
//  * All dense math runs through one bf16 WMMA GEMM kernel
//    (v_wmma_f32_16x16x32_bf16), wave32.
//  * Wave tile = 16(M) x 64(N) (4 WMMA tiles, shared A fragment) when N%64==0;
//    falls back to 16x16 for the N=50000 retrieval GEMM.
//  * f32 inputs/weights are converted to bf16 (RNE) once; each weight is used
//    once per forward pass so conversion traffic is negligible vs. the 8x
//    denser math than V_WMMA_F32_16X16X4_F32.
// ============================================================================

typedef __attribute__((ext_vector_type(16))) __bf16 v16bf;
typedef __attribute__((ext_vector_type(8)))  float  v8f;
typedef __attribute__((ext_vector_type(4)))  unsigned int u32x4;

union FragAB { u32x4 u[2]; v16bf bf; };

__device__ __forceinline__ uint16_t f2bfu(float f) {
  union { float f; unsigned u; } x; x.f = f;
  unsigned r = x.u + 0x7FFFu + ((x.u >> 16) & 1u);   // round-to-nearest-even
  return (uint16_t)(r >> 16);
}

__device__ __forceinline__ float act_apply(float v, int act) {
  if (act == 1) return fmaxf(v, 0.f);                                  // relu
  if (act == 2) return 0.5f * v * (1.0f + erff(v * 0.70710678118f));   // exact gelu
  if (act == 3) return 1.f / (1.f + expf(-v));                         // sigmoid
  return v;
}

__device__ __forceinline__ float blockReduceSum(float v, float* red) {
  int t = threadIdx.x;
  red[t] = v; __syncthreads();
  for (int s = 128; s > 0; s >>= 1) { if (t < s) red[t] += red[t + s]; __syncthreads(); }
  float r = red[0]; __syncthreads();
  return r;
}

// ---------------------------------------------------------------------------
// WMMA GEMM:  out[M,N] = act(A[M,K] @ W[N,K]^T + bias[N])
// A, W in bf16 (row-major, K contiguous). Block = 256 thr = 8 waves.
// Wave tile = 16(M) x 16*NT(N); block tile = 128(M) x 16*NT(N).
// grid = (N/(16*NT), M/128). K must be a multiple of 32.
// ---------------------------------------------------------------------------
template <int NT>
__global__ __launch_bounds__(256)
void k_gemm(const uint16_t* __restrict__ A, const uint16_t* __restrict__ W,
            const float* __restrict__ bias, float* __restrict__ outF,
            uint16_t* __restrict__ outB, int M, int N, int K, int act)
{
  const int lane = threadIdx.x & 31;
  const int wave = threadIdx.x >> 5;
  const int mr   = blockIdx.y * 128 + wave * 16;
  const int nc0  = blockIdx.x * (16 * NT);
  if (mr >= M) return;
  const int r    = lane & 15;
  const int half = lane >> 4;

  const uint16_t* Arow = A + (size_t)(mr + r) * K;   // A frag: row = mr + (lane&15)
  const uint16_t* Wc[NT];
  #pragma unroll
  for (int t = 0; t < NT; ++t) Wc[t] = W + (size_t)(nc0 + t * 16 + r) * K;

  v8f acc[NT];
  #pragma unroll
  for (int t = 0; t < NT; ++t) acc[t] = (v8f){};

  for (int kb = 0; kb < K; kb += 32) {
    // A 16x32 bf16 layout: lane<16 -> K {kb..kb+7, kb+16..kb+23}; lane>=16 -> +8
    FragAB a;
    const int ka = kb + half * 8;
    a.u[0] = *(const u32x4*)(Arow + ka);
    a.u[1] = *(const u32x4*)(Arow + ka + 16);
    // B 32x16 layout: lane<16 -> K {kb..kb+7, kb+16..kb+23}; lane>=16 -> +8
    const int kw = kb + half * 8;
    FragAB b[NT];
    #pragma unroll
    for (int t = 0; t < NT; ++t) {
      b[t].u[0] = *(const u32x4*)(Wc[t] + kw);
      b[t].u[1] = *(const u32x4*)(Wc[t] + kw + 16);
    }
    #pragma unroll
    for (int t = 0; t < NT; ++t)
      acc[t] = __builtin_amdgcn_wmma_f32_16x16x32_bf16(false, a.bf, false, b[t].bf,
                                                       (short)0, acc[t], false, false);
  }

  #pragma unroll
  for (int t = 0; t < NT; ++t) {
    const int col = nc0 + t * 16 + r;
    const float bv = bias ? bias[col] : 0.f;
    #pragma unroll
    for (int i = 0; i < 8; ++i) {
      const int row = mr + half * 8 + i;          // C/D layout: vgpr i, half -> M
      float v = act_apply(acc[t][i] + bv, act);
      const size_t o = (size_t)row * N + col;
      if (outF) outF[o] = v;
      if (outB) outB[o] = f2bfu(v);
    }
  }
}

// ---------------------------------------------------------------------------
// Elementwise / row kernels
// ---------------------------------------------------------------------------
__global__ __launch_bounds__(256)
void k_cvt(const float* __restrict__ s, uint16_t* __restrict__ d, size_t n) {
  size_t i = (size_t)blockIdx.x * 256 + threadIdx.x;
  size_t st = (size_t)gridDim.x * 256;
  for (; i < n; i += st) d[i] = f2bfu(s[i]);
}

__global__ __launch_bounds__(256)
void k_l2norm(const float* __restrict__ x, uint16_t* __restrict__ out, int cols) {
  const int row = blockIdx.x;
  const float* xr = x + (size_t)row * cols;
  __shared__ float red[256];
  float s = 0.f;
  for (int c = threadIdx.x; c < cols; c += 256) { float v = xr[c]; s += v * v; }
  float nrm = sqrtf(blockReduceSum(s, red));
  float inv = 1.f / fmaxf(nrm, 1e-12f);
  uint16_t* d = out + (size_t)row * cols;
  for (int c = threadIdx.x; c < cols; c += 256) d[c] = f2bfu(xr[c] * inv);
}

// y = LN(x (+res)) * g + b ; optional relu ; optional (+post) afterwards
__global__ __launch_bounds__(256)
void k_rowln(const float* __restrict__ x, const float* __restrict__ res,
             const float* __restrict__ g, const float* __restrict__ b,
             const float* __restrict__ post, float* __restrict__ outF,
             uint16_t* __restrict__ outB, int cols, int relu)
{
  const int row = blockIdx.x;
  const float* xr = x + (size_t)row * cols;
  const float* rr = res ? res + (size_t)row * cols : nullptr;
  __shared__ float red[256];
  float s = 0.f;
  for (int c = threadIdx.x; c < cols; c += 256) s += xr[c] + (rr ? rr[c] : 0.f);
  const float mean = blockReduceSum(s, red) / cols;
  float s2 = 0.f;
  for (int c = threadIdx.x; c < cols; c += 256) {
    float v = xr[c] + (rr ? rr[c] : 0.f) - mean; s2 += v * v;
  }
  const float rstd = rsqrtf(blockReduceSum(s2, red) / cols + 1e-5f);
  for (int c = threadIdx.x; c < cols; c += 256) {
    float v = (xr[c] + (rr ? rr[c] : 0.f) - mean) * rstd * g[c] + b[c];
    if (relu) v = fmaxf(v, 0.f);
    const size_t o = (size_t)row * cols + c;
    if (post) v += post[o];
    if (outF) outF[o] = v;
    if (outB) outB[o] = f2bfu(v);
  }
}

// Per-row top-10 over 50000 columns; thread-local sorted lists + LDS tree merge.
#define TK 10
__global__ __launch_bounds__(256)
void k_topk(const float* __restrict__ sims, int* __restrict__ idxOut, int cols) {
  const int row = blockIdx.x;
  const float* sr = sims + (size_t)row * cols;
  float bv[TK]; int bi[TK];
  #pragma unroll
  for (int k = 0; k < TK; ++k) { bv[k] = -3.4e38f; bi[k] = 0; }
  for (int c = threadIdx.x; c < cols; c += 256) {
    float v = sr[c];
    if (v > bv[TK - 1]) {
      int j = TK - 1;
      while (j > 0 && bv[j - 1] < v) { bv[j] = bv[j - 1]; bi[j] = bi[j - 1]; --j; }
      bv[j] = v; bi[j] = c;
    }
  }
  __shared__ float sv[256 * TK];
  __shared__ int   si[256 * TK];
  const int t = threadIdx.x;
  #pragma unroll
  for (int k = 0; k < TK; ++k) { sv[t * TK + k] = bv[k]; si[t * TK + k] = bi[k]; }
  __syncthreads();
  for (int stride = 128; stride > 0; stride >>= 1) {
    if (t < stride) {
      float mv[TK]; int mi[TK];
      float* av = sv + t * TK;            int* ai = si + t * TK;
      float* bw = sv + (t + stride) * TK; int* bj = si + (t + stride) * TK;
      int i = 0, j = 0;
      #pragma unroll
      for (int k = 0; k < TK; ++k) {
        if (av[i] >= bw[j]) { mv[k] = av[i]; mi[k] = ai[i]; ++i; }
        else                { mv[k] = bw[j]; mi[k] = bj[j]; ++j; }
      }
      #pragma unroll
      for (int k = 0; k < TK; ++k) { av[k] = mv[k]; ai[k] = mi[k]; }
    }
    __syncthreads();
  }
  if (t < TK) idxOut[row * TK + t] = si[t];
}

__global__ __launch_bounds__(256)
void k_gather(const float* __restrict__ bank, const int* __restrict__ idx,
              uint16_t* __restrict__ dst, int C) {
  const int row = blockIdx.x;              // b*10 + j
  const int src = idx[row];
  const float* s = bank + (size_t)src * C;
  uint16_t* d = dst + (size_t)row * C;
  for (int c = threadIdx.x; c < C; c += 256) d[c] = f2bfu(s[c]);
}

// Assemble Kc/Vc (256 x 11 x 768) bf16: row j==0 -> proto[b]; else nb[b,j-1]
__global__ __launch_bounds__(256)
void k_concat_pn(const float* __restrict__ proto, const float* __restrict__ nb,
                 uint16_t* __restrict__ dst) {
  const int row = blockIdx.x;              // 0..2815
  const int b = row / 11, j = row % 11;
  const float* s = (j == 0) ? (proto + (size_t)b * 768)
                            : (nb + ((size_t)(b * 10 + j - 1)) * 768);
  uint16_t* d = dst + (size_t)row * 768;
  for (int c = threadIdx.x; c < 768; c += 256) d[c] = f2bfu(s[c]);
}

// prototype classification: cls = argmax(l2n(avail) . l2n(proto_k)); copy proto
__global__ __launch_bounds__(256)
void k_proto(const float* __restrict__ avail, const float* __restrict__ protos,
             float* __restrict__ proto_out) {
  const int b = blockIdx.x;
  const int t = threadIdx.x;
  __shared__ float red[256];
  __shared__ float sims4[4];
  __shared__ int cls;
  float s = 0.f;
  for (int c = t; c < 768; c += 256) { float v = avail[(size_t)b * 768 + c]; s += v * v; }
  const float an = fmaxf(sqrtf(blockReduceSum(s, red)), 1e-12f);
  for (int k = 0; k < 4; ++k) {
    float d = 0.f, n2 = 0.f;
    for (int c = t; c < 768; c += 256) {
      float p = protos[k * 768 + c];
      d += avail[(size_t)b * 768 + c] * p; n2 += p * p;
    }
    d  = blockReduceSum(d, red);
    n2 = blockReduceSum(n2, red);
    if (t == 0) sims4[k] = d / (an * fmaxf(sqrtf(n2), 1e-12f));
    __syncthreads();
  }
  if (t == 0) {
    int best = 0;
    for (int k = 1; k < 4; ++k) if (sims4[k] > sims4[best]) best = k;
    cls = best;
  }
  __syncthreads();
  for (int c = t; c < 768; c += 256)
    proto_out[(size_t)b * 768 + c] = protos[cls * 768 + c];
}

// pnfc attention: Lq=1, Lk=11, H=4, hd=192
__global__ __launch_bounds__(256)
void k_attn11(const float* __restrict__ Qh, const float* __restrict__ Kh,
              const float* __restrict__ Vh, float* __restrict__ O) {
  const int b = blockIdx.x;
  const int t = threadIdx.x;
  __shared__ float s[4][11];
  if (t < 44) {
    const int h = t / 11, j = t % 11;
    const float* q = Qh + (size_t)b * 768 + h * 192;
    const float* k = Kh + ((size_t)(b * 11 + j)) * 768 + h * 192;
    float d = 0.f;
    for (int e = 0; e < 192; ++e) d += q[e] * k[e];
    s[h][j] = d * 0.07216878364870323f;   // 1/sqrt(192)
  }
  __syncthreads();
  if (t < 4) {
    float m = -3.4e38f;
    for (int j = 0; j < 11; ++j) m = fmaxf(m, s[t][j]);
    float sum = 0.f;
    for (int j = 0; j < 11; ++j) { float e = expf(s[t][j] - m); s[t][j] = e; sum += e; }
    const float inv = 1.f / sum;
    for (int j = 0; j < 11; ++j) s[t][j] *= inv;
  }
  __syncthreads();
  for (int c = t; c < 768; c += 256) {
    const int h = c / 192;
    float acc = 0.f;
    for (int j = 0; j < 11; ++j) acc += s[h][j] * Vh[((size_t)(b * 11 + j)) * 768 + c];
    O[(size_t)b * 768 + c] = acc;
  }
}

// fusion transformer self-attention: L=8, H=4, hd=192 on packed qkv(2048x2304)
__global__ __launch_bounds__(256)
void k_attn8(const float* __restrict__ qkv, float* __restrict__ O) {
  const int b = blockIdx.x;
  const int t = threadIdx.x;
  __shared__ float s[4][8][8];
  {
    const int h = t >> 6, i = (t >> 3) & 7, j = t & 7;
    const float* q = qkv + ((size_t)(b * 8 + i)) * 2304 + h * 192;
    const float* k = qkv + ((size_t)(b * 8 + j)) * 2304 + 768 + h * 192;
    float d = 0.f;
    for (int e = 0; e < 192; ++e) d += q[e] * k[e];
    s[h][i][j] = d * 0.07216878364870323f;
  }
  __syncthreads();
  if (t < 32) {
    const int h = t >> 3, i = t & 7;
    float m = -3.4e38f;
    for (int j = 0; j < 8; ++j) m = fmaxf(m, s[h][i][j]);
    float sum = 0.f;
    for (int j = 0; j < 8; ++j) { float e = expf(s[h][i][j] - m); s[h][i][j] = e; sum += e; }
    const float inv = 1.f / sum;
    for (int j = 0; j < 8; ++j) s[h][i][j] *= inv;
  }
  __syncthreads();
  for (int e = t; e < 8 * 768; e += 256) {
    const int i = e / 768, c = e % 768, h = c / 192;
    float acc = 0.f;
    for (int j = 0; j < 8; ++j)
      acc += s[h][i][j] * qkv[((size_t)(b * 8 + j)) * 2304 + 1536 + c];
    O[((size_t)(b * 8 + i)) * 768 + c] = acc;
  }
}

// build x = concat([seq, pivot], tokens) : (256,8,768), f32 + bf16
__global__ __launch_bounds__(256)
void k_xbuild(const float* __restrict__ seq, const float* __restrict__ pivot,
              float* __restrict__ xF, uint16_t* __restrict__ xB) {
  const int row = blockIdx.x;              // 0..2047
  const int b = row >> 3, tok = row & 7;
  const float* s = (tok < 4) ? (seq + (size_t)b * 3072 + tok * 768)
                             : (pivot + (size_t)b * 3072 + (tok - 4) * 768);
  float* df = xF + (size_t)row * 768;
  uint16_t* db = xB + (size_t)row * 768;
  for (int c = threadIdx.x; c < 768; c += 256) { float v = s[c]; df[c] = v; db[c] = f2bfu(v); }
}

__global__ __launch_bounds__(256)
void k_seqcopy(const float* __restrict__ tfout, float* __restrict__ seq) {
  const int b = blockIdx.x;
  for (int e = threadIdx.x; e < 3072; e += 256)
    seq[(size_t)b * 3072 + e] = tfout[(size_t)b * 6144 + e];   // tokens 0..3
}

__global__ __launch_bounds__(256)
void k_pvmean(const float* __restrict__ tfout, float* __restrict__ outF,
              uint16_t* __restrict__ outB) {
  const int b = blockIdx.x;
  for (int c = threadIdx.x; c < 768; c += 256) {
    float s = 0.f;
    for (int p = 0; p < 4; ++p) s += tfout[(size_t)b * 6144 + 3072 + p * 768 + c];
    s *= 0.25f;
    outF[(size_t)b * 768 + c] = s;
    outB[(size_t)b * 768 + c] = f2bfu(s);
  }
}

__global__ __launch_bounds__(256)
void k_alpha(const float* __restrict__ h1, const float* __restrict__ w2,
             const float* __restrict__ b2, float* __restrict__ al) {
  const int b = threadIdx.x;               // one block of 256
  float s = 0.f;
  for (int e = 0; e < 384; ++e) s += h1[(size_t)b * 384 + e] * w2[e];
  s += b2[0];
  al[b] = 1.f / (1.f + expf(-s));
}

// pivot[b,p,:] = LN(al[b]*pivot + (1-al[b])*pv[b,p,:]) * g + b
__global__ __launch_bounds__(256)
void k_pivot_ln(float* __restrict__ pivot, const float* __restrict__ tfout,
                const float* __restrict__ al, const float* __restrict__ g,
                const float* __restrict__ b) {
  const int row = blockIdx.x;              // 0..1023 = b*4+p
  const int bI = row >> 2, p = row & 3;
  const float a = al[bI];
  float* dst = pivot + (size_t)row * 768;
  const float* pv = tfout + (size_t)bI * 6144 + 3072 + p * 768;
  __shared__ float mix[768];
  __shared__ float red[256];
  for (int c = threadIdx.x; c < 768; c += 256)
    mix[c] = a * dst[c] + (1.f - a) * pv[c];
  __syncthreads();
  float s = 0.f;
  for (int c = threadIdx.x; c < 768; c += 256) s += mix[c];
  const float mean = blockReduceSum(s, red) * (1.f / 768.f);
  float s2 = 0.f;
  for (int c = threadIdx.x; c < 768; c += 256) { float v = mix[c] - mean; s2 += v * v; }
  const float rstd = rsqrtf(blockReduceSum(s2, red) * (1.f / 768.f) + 1e-5f);
  for (int c = threadIdx.x; c < 768; c += 256)
    dst[c] = (mix[c] - mean) * rstd * g[c] + b[c];
}

__global__ __launch_bounds__(256)
void k_concat3(const float* __restrict__ t, const float* __restrict__ v,
               const float* __restrict__ a, uint16_t* __restrict__ dst) {
  const int b = blockIdx.x;
  for (int c = threadIdx.x; c < 2304; c += 256) {
    float x = (c < 768) ? t[(size_t)b * 768 + c]
            : (c < 1536) ? v[(size_t)b * 768 + c - 768]
                         : a[(size_t)b * 768 + c - 1536];
    dst[(size_t)b * 2304 + c] = f2bfu(x);
  }
}

__global__ __launch_bounds__(256)
void k_avg3(const float* __restrict__ t, const float* __restrict__ v,
            const float* __restrict__ a, float* __restrict__ o, size_t n) {
  size_t i = (size_t)blockIdx.x * 256 + threadIdx.x;
  size_t st = (size_t)gridDim.x * 256;
  for (; i < n; i += st) o[i] = (t[i] + v[i] + a[i]) * (1.f / 3.f);
}

__global__ __launch_bounds__(256)
void k_fuse(const float* __restrict__ gate, const float* __restrict__ pf,
            const float* __restrict__ raw, float* __restrict__ fused, size_t n) {
  size_t i = (size_t)blockIdx.x * 256 + threadIdx.x;
  size_t st = (size_t)gridDim.x * 256;
  for (; i < n; i += st) { float g = gate[i]; fused[i] = g * pf[i] + (1.f - g) * raw[i]; }
}

__global__ __launch_bounds__(256)
void k_cls(const float* __restrict__ fused, const float* __restrict__ w,
           const float* __restrict__ bias, float* __restrict__ out) {
  const int n = blockIdx.x;                // 8
  const int b = threadIdx.x;               // 256
  float s = 0.f;
  for (int e = 0; e < 768; ++e) s += fused[(size_t)b * 768 + e] * w[n * 768 + e];
  out[(size_t)b * 8 + n] = s + bias[n];
}

// ---------------------------------------------------------------------------
// Host-side plumbing
// ---------------------------------------------------------------------------
struct Bump {
  char* p;
  template <typename T> T* take(size_t n) {
    T* r = (T*)p;
    size_t bytes = (n * sizeof(T) + 255) & ~(size_t)255;
    p += bytes;
    return r;
  }
};

static inline void cvtW(const float* s, uint16_t* d, size_t n, hipStream_t st) {
  unsigned blocks = (unsigned)((n + 2047) / 2048);
  if (blocks > 4096) blocks = 4096;
  if (blocks == 0) blocks = 1;
  k_cvt<<<dim3(blocks), dim3(256), 0, st>>>(s, d, n);
}
static inline void gemmW(const uint16_t* A, const uint16_t* W, const float* bias,
                         float* outF, uint16_t* outB, int M, int N, int K, int act,
                         hipStream_t st) {
  if (N % 64 == 0)
    k_gemm<4><<<dim3(N / 64, M / 128), dim3(256), 0, st>>>(A, W, bias, outF, outB, M, N, K, act);
  else
    k_gemm<1><<<dim3(N / 16, M / 128), dim3(256), 0, st>>>(A, W, bias, outF, outB, M, N, K, act);
}
static inline void rowlnW(const float* x, const float* res, const float* g,
                          const float* b, const float* post, float* outF,
                          uint16_t* outB, int rows, int cols, int relu, hipStream_t st) {
  k_rowln<<<dim3(rows), dim3(256), 0, st>>>(x, res, g, b, post, outF, outB, cols, relu);
}

// Parameter leaf indices (jax pytree flattening: dict keys alphabetical).
enum {
  IN_TEXT = 0, IN_BANK_T = 1, IN_BANK_I = 2, IN_BANK_A = 3,
  P_CLS_B = 4, P_CLS_W = 5,
  F_AA = 6,   /* b1,b2,w1,w2 */ F_ASG_B = 10, F_ASG_W = 11,
  F_AT = 12,  F_AV = 16,
  F_GATE_B1 = 20, F_GATE_B2 = 21, F_GATE_W1 = 22, F_GATE_W2 = 23,
  F_INFUSE_B = 24, F_INFUSE_W = 25,
  F_LNA_B = 26, F_LNA_G = 27, F_LNP_B = 28, F_LNP_G = 29,
  F_LNT_B = 30, F_LNT_G = 31, F_LNV_B = 32, F_LNV_G = 33,
  F_OUT_B = 34, F_OUT_W = 35,
  F_TF0 = 36,  /* 6 layers x 12: in_b,in_w,out_b,out_w,b1,b2,n1b,n1g,n2b,n2g,w1,w2 */
  F_TSG_B = 108, F_TSG_W = 109, F_VSG_B = 110, F_VSG_W = 111,
  PN_APROJ = 112, /* b,bt,g,w */
  PN_IN_B = 116, PN_IN_W = 117, PN_OUT_B = 118, PN_OUT_W = 119,
  PN_FB1 = 120, PN_FB2 = 121, PN_FW1 = 122, PN_FW2 = 123,
  PN_IPROJ = 124,
  PN_LN1B = 128, PN_LN1G = 129, PN_LN2B = 130, PN_LN2G = 131,
  PN_TPROJ = 132,
  P_PROTO = 136,
  P_TPROJ = 137 /* b,bt,g,w */
};

extern "C" void kernel_launch(void* const* d_in, const int* in_sizes, int n_in,
                              void* d_out, int out_size, void* d_ws, size_t ws_size,
                              hipStream_t stream) {
  (void)in_sizes; (void)n_in; (void)out_size; (void)ws_size;
  auto F = [&](int i) { return (const float*)d_in[i]; };

  const int B = 256, NB = 50000, TOPK = 10, TD = 768, ID = 1024, AD = 512, D = 768;
  const float* text = F(IN_TEXT);

  Bump ws{(char*)d_ws};
  int*      idx    = ws.take<int>(B * TOPK);
  uint16_t* WB     = ws.take<uint16_t>(768 * 3072);      // shared weight staging
  uint16_t* tBraw  = ws.take<uint16_t>((size_t)B * TD);  // text bf16 (raw)
  uint16_t* pnInW  = ws.take<uint16_t>(2304 * 768);      // pnfc attn/ffn (reused 2x)
  uint16_t* pnOutW = ws.take<uint16_t>(768 * 768);
  uint16_t* pnFw1  = ws.take<uint16_t>(3072 * 768);
  uint16_t* pnFw2  = ws.take<uint16_t>(768 * 3072);
  char* arenaStart = ws.p;

  cvtW(text, tBraw, (size_t)B * TD, stream);
  cvtW(F(PN_IN_W),  pnInW,  (size_t)2304 * 768, stream);
  cvtW(F(PN_OUT_W), pnOutW, (size_t)768 * 768,  stream);
  cvtW(F(PN_FW1),   pnFw1,  (size_t)3072 * 768, stream);
  cvtW(F(PN_FW2),   pnFw2,  (size_t)768 * 3072, stream);

  // ---- Stage 1: retrieval (sims + top-k); arena reclaimed afterwards ----
  {
    Bump a{arenaStart};
    uint16_t* tN    = a.take<uint16_t>((size_t)B * TD);
    uint16_t* bankN = a.take<uint16_t>((size_t)NB * TD);
    float*    sims  = a.take<float>((size_t)B * NB);
    k_l2norm<<<dim3(B), dim3(256), 0, stream>>>(text, tN, TD);
    k_l2norm<<<dim3(NB), dim3(256), 0, stream>>>(F(IN_BANK_T), bankN, TD);
    gemmW(tN, bankN, nullptr, sims, nullptr, B, NB, TD, 0, stream);   // 19.7 GF bf16 WMMA
    k_topk<<<dim3(B), dim3(256), 0, stream>>>(sims, idx, NB);
  }

  // ---- Stage 2+: everything else ----
  Bump a{arenaStart};
  uint16_t* nbT   = a.take<uint16_t>((size_t)2560 * TD);
  uint16_t* nbI   = a.take<uint16_t>((size_t)2560 * ID);
  uint16_t* nbA   = a.take<uint16_t>((size_t)2560 * AD);
  float* tmp2560  = a.take<float>((size_t)2560 * D);
  float* tmp256   = a.take<float>((size_t)B * D);
  float* avail    = a.take<float>((size_t)B * D);
  uint16_t* availB = a.take<uint16_t>((size_t)B * D);
  float* avail_nb = a.take<float>((size_t)2560 * D);
  float* vis_in   = a.take<float>((size_t)2560 * D);
  float* aud_in   = a.take<float>((size_t)2560 * D);
  float* text_c   = a.take<float>((size_t)B * D);
  float* proto    = a.take<float>((size_t)B * D);
  uint16_t* KcB   = a.take<uint16_t>((size_t)2816 * D);
  uint16_t* VcB   = a.take<uint16_t>((size_t)2816 * D);
  float* Qh       = a.take<float>((size_t)B * D);
  float* Kh       = a.take<float>((size_t)2816 * D);
  float* Vh       = a.take<float>((size_t)2816 * D);
  float* O256     = a.take<float>((size_t)B * D);
  uint16_t* O256B = a.take<uint16_t>((size_t)B * D);
  float* aBuf     = a.take<float>((size_t)B * D);
  float* oBuf     = a.take<float>((size_t)B * D);
  uint16_t* oB    = a.take<uint16_t>((size_t)B * D);
  uint16_t* ffhB  = a.take<uint16_t>((size_t)2048 * 3072);
  float* ff2      = a.take<float>((size_t)2048 * D);
  float* visual   = a.take<float>((size_t)B * D);
  float* audio    = a.take<float>((size_t)B * D);
  float* tF = a.take<float>((size_t)B * D);  uint16_t* tLB = a.take<uint16_t>((size_t)B * D);
  float* vF = a.take<float>((size_t)B * D);  uint16_t* vLB = a.take<uint16_t>((size_t)B * D);
  float* aF = a.take<float>((size_t)B * D);  uint16_t* aLB = a.take<uint16_t>((size_t)B * D);
  uint16_t* cat3B = a.take<uint16_t>((size_t)B * 2304);
  float* raw      = a.take<float>((size_t)B * D);
  float* tseq     = a.take<float>((size_t)B * 3072);
  float* vseq     = a.take<float>((size_t)B * 3072);
  float* aseq     = a.take<float>((size_t)B * 3072);
  float* pivot    = a.take<float>((size_t)B * 3072);
  float* xF       = a.take<float>((size_t)2048 * D);
  uint16_t* xB    = a.take<uint16_t>((size_t)2048 * D);
  float* qkv      = a.take<float>((size_t)2048 * 2304);
  float* O2048    = a.take<float>((size_t)2048 * D);
  uint16_t* O2048B = a.take<uint16_t>((size_t)2048 * D);
  float* attnO    = a.take<float>((size_t)2048 * D);
  float* x1F      = a.take<float>((size_t)2048 * D);
  uint16_t* x1B   = a.take<uint16_t>((size_t)2048 * D);
  float* tfout    = a.take<float>((size_t)2048 * D);
  float* pvmF     = a.take<float>((size_t)B * D);
  uint16_t* pvmB  = a.take<uint16_t>((size_t)B * D);
  float* h1       = a.take<float>((size_t)B * 384);
  float* al       = a.take<float>(B);
  uint16_t* pivotB = a.take<uint16_t>((size_t)B * 3072);
  float* pf       = a.take<float>((size_t)B * D);
  uint16_t* pfB   = a.take<uint16_t>((size_t)B * D);
  uint16_t* g1B   = a.take<uint16_t>((size_t)B * 384);
  float* gate     = a.take<float>((size_t)B * D);
  float* fused    = a.take<float>((size_t)B * D);

  // gathers (f32 bank -> bf16 neighbor tiles)
  k_gather<<<dim3(2560), dim3(256), 0, stream>>>(F(IN_BANK_T), idx, nbT, TD);
  k_gather<<<dim3(2560), dim3(256), 0, stream>>>(F(IN_BANK_I), idx, nbI, ID);
  k_gather<<<dim3(2560), dim3(256), 0, stream>>>(F(IN_BANK_A), idx, nbA, AD);

  // projc(text & nb_t, tproj) -> avail / avail_nb
  cvtW(F(PN_TPROJ + 3), WB, (size_t)D * TD, stream);
  gemmW(tBraw, WB, F(PN_TPROJ + 0), tmp256, nullptr, B, D, TD, 0, stream);
  rowlnW(tmp256, nullptr, F(PN_TPROJ + 2), F(PN_TPROJ + 1), nullptr, avail, availB, B, D, 1, stream);
  gemmW(nbT, WB, F(PN_TPROJ + 0), tmp2560, nullptr, 2560, D, TD, 0, stream);
  rowlnW(tmp2560, nullptr, F(PN_TPROJ + 2), F(PN_TPROJ + 1), nullptr, avail_nb, nullptr, 2560, D, 1, stream);
  // projc(nb_i, iproj) ; projc(nb_a, aproj)
  cvtW(F(PN_IPROJ + 3), WB, (size_t)D * ID, stream);
  gemmW(nbI, WB, F(PN_IPROJ + 0), tmp2560, nullptr, 2560, D, ID, 0, stream);
  rowlnW(tmp2560, nullptr, F(PN_IPROJ + 2), F(PN_IPROJ + 1), nullptr, vis_in, nullptr, 2560, D, 1, stream);
  cvtW(F(PN_APROJ + 3), WB, (size_t)D * AD, stream);
  gemmW(nbA, WB, F(PN_APROJ + 0), tmp2560, nullptr, 2560, D, AD, 0, stream);
  rowlnW(tmp2560, nullptr, F(PN_APROJ + 2), F(PN_APROJ + 1), nullptr, aud_in, nullptr, 2560, D, 1, stream);
  // projc(text, text_proj) -> text_c
  cvtW(F(P_TPROJ + 3), WB, (size_t)D * TD, stream);
  gemmW(tBraw, WB, F(P_TPROJ + 0), tmp256, nullptr, B, D, TD, 0, stream);
  rowlnW(tmp256, nullptr, F(P_TPROJ + 2), F(P_TPROJ + 1), nullptr, text_c, nullptr, B, D, 1, stream);

  // prototype selection
  k_proto<<<dim3(B), dim3(256), 0, stream>>>(avail, F(P_PROTO), proto);

  // complete(): shared Q/K across both modalities
  k_concat_pn<<<dim3(2816), dim3(256), 0, stream>>>(proto, avail_nb, KcB);
  gemmW(availB, pnInW,              F(PN_IN_B),        Qh, nullptr, B,    D, D, 0, stream);
  gemmW(KcB,    pnInW + 768 * 768,  F(PN_IN_B) + 768,  Kh, nullptr, 2816, D, D, 0, stream);

  const float* missArr[2] = { vis_in, aud_in };
  float* resArr[2] = { visual, audio };
  for (int m = 0; m < 2; ++m) {
    k_concat_pn<<<dim3(2816), dim3(256), 0, stream>>>(proto, missArr[m], VcB);
    gemmW(VcB, pnInW + 2 * 768 * 768, F(PN_IN_B) + 1536, Vh, nullptr, 2816, D, D, 0, stream);
    k_attn11<<<dim3(B), dim3(256), 0, stream>>>(Qh, Kh, Vh, O256);
    cvtW(O256, O256B, (size_t)B * D, stream);
    gemmW(O256B, pnOutW, F(PN_OUT_B), aBuf, nullptr, B, D, D, 0, stream);
    rowlnW(aBuf, proto, F(PN_LN1G), F(PN_LN1B), nullptr, oBuf, oB, B, D, 0, stream);
    gemmW(oB, pnFw1, F(PN_FB1), nullptr, ffhB, B, 3072, D, 1 /*relu*/, stream);
    gemmW(ffhB, pnFw2, F(PN_FB2), ff2, nullptr, B, D, 3072, 0, stream);
    rowlnW(ff2, oBuf, F(PN_LN2G), F(PN_LN2B), proto /*+proto*/, resArr[m], nullptr, B, D, 0, stream);
  }

  // fusion front-end
  rowlnW(text_c, nullptr, F(F_LNT_G), F(F_LNT_B), nullptr, tF, tLB, B, D, 0, stream);
  rowlnW(visual, nullptr, F(F_LNV_G), F(F_LNV_B), nullptr, vF, vLB, B, D, 0, stream);
  rowlnW(audio,  nullptr, F(F_LNA_G), F(F_LNA_B), nullptr, aF, aLB, B, D, 0, stream);
  k_concat3<<<dim3(B), dim3(256), 0, stream>>>(tF, vF, aF, cat3B);
  cvtW(F(F_INFUSE_W), WB, (size_t)D * 2304, stream);
  gemmW(cat3B, WB, F(F_INFUSE_B), raw, nullptr, B, D, 2304, 0, stream);
  cvtW(F(F_TSG_W), WB, (size_t)3072 * D, stream);
  gemmW(tLB, WB, F(F_TSG_B), tseq, nullptr, B, 3072, D, 0, stream);
  cvtW(F(F_VSG_W), WB, (size_t)3072 * D, stream);
  gemmW(vLB, WB, F(F_VSG_B), vseq, nullptr, B, 3072, D, 0, stream);
  cvtW(F(F_ASG_W), WB, (size_t)3072 * D, stream);
  gemmW(aLB, WB, F(F_ASG_B), aseq, nullptr, B, 3072, D, 0, stream);
  k_avg3<<<dim3(1024), dim3(256), 0, stream>>>(tseq, vseq, aseq, pivot, (size_t)B * 3072);

  // transformer fusion loop: NL=2 x ('v','a','t')
  float* seqs[3] = { vseq, aseq, tseq };
  const int anetBase[3] = { F_AV, F_AA, F_AT };   // anet leaves: b1,b2,w1,w2
  for (int li = 0; li < 2; ++li) {
    for (int si = 0; si < 3; ++si) {
      const int tb = F_TF0 + (li * 3 + si) * 12;
      float* seq = seqs[si];
      k_xbuild<<<dim3(2048), dim3(256), 0, stream>>>(seq, pivot, xF, xB);
      cvtW(F(tb + 1), WB, (size_t)2304 * D, stream);                 // in_w
      gemmW(xB, WB, F(tb + 0), qkv, nullptr, 2048, 2304, D, 0, stream);
      k_attn8<<<dim3(B), dim3(256), 0, stream>>>(qkv, O2048);
      cvtW(O2048, O2048B, (size_t)2048 * D, stream);
      cvtW(F(tb + 3), WB, (size_t)D * D, stream);                    // out_w
      gemmW(O2048B, WB, F(tb + 2), attnO, nullptr, 2048, D, D, 0, stream);
      rowlnW(xF, attnO, F(tb + 7), F(tb + 6), nullptr, x1F, x1B, 2048, D, 0, stream);
      cvtW(F(tb + 10), WB, (size_t)3072 * D, stream);                // w1
      gemmW(x1B, WB, F(tb + 4), nullptr, ffhB, 2048, 3072, D, 2 /*gelu*/, stream);
      cvtW(F(tb + 11), WB, (size_t)D * 3072, stream);                // w2
      gemmW(ffhB, WB, F(tb + 5), ff2, nullptr, 2048, D, 3072, 0, stream);
      rowlnW(x1F, ff2, F(tb + 9), F(tb + 8), nullptr, tfout, nullptr, 2048, D, 0, stream);
      k_seqcopy<<<dim3(B), dim3(256), 0, stream>>>(tfout, seq);
      k_pvmean<<<dim3(B), dim3(256), 0, stream>>>(tfout, pvmF, pvmB);
      const int ab = anetBase[si];
      cvtW(F(ab + 2), WB, (size_t)384 * D, stream);                  // anet w1
      gemmW(pvmB, WB, F(ab + 0), h1, nullptr, B, 384, D, 2 /*gelu*/, stream);
      k_alpha<<<dim3(1), dim3(256), 0, stream>>>(h1, F(ab + 3), F(ab + 1), al);
      k_pivot_ln<<<dim3(1024), dim3(256), 0, stream>>>(pivot, tfout, al,
                                                       F(F_LNP_G), F(F_LNP_B));
    }
  }

  // head: pf, gate, fuse, classify
  cvtW(pivot, pivotB, (size_t)B * 3072, stream);
  cvtW(F(F_OUT_W), WB, (size_t)D * 3072, stream);
  gemmW(pivotB, WB, F(F_OUT_B), pf, pfB, B, D, 3072, 2 /*gelu*/, stream);
  cvtW(F(F_GATE_W1), WB, (size_t)384 * D, stream);
  gemmW(pfB, WB, F(F_GATE_B1), nullptr, g1B, B, 384, D, 2 /*gelu*/, stream);
  cvtW(F(F_GATE_W2), WB, (size_t)D * 384, stream);
  gemmW(g1B, WB, F(F_GATE_B2), gate, nullptr, B, D, 384, 3 /*sigmoid*/, stream);
  k_fuse<<<dim3(768), dim3(256), 0, stream>>>(gate, pf, raw, fused, (size_t)B * D);
  k_cls<<<dim3(8), dim3(256), 0, stream>>>(fused, F(P_CLS_W), F(P_CLS_B), (float*)d_out);
}
